// RelPosMHAXL_24704651887076
// MI455X (gfx1250) — compile-verified, matrix operand order checked by reading, not run on hardware
//
#include <hip/hip_runtime.h>
#include <hip/hip_bf16.h>
#include <cstdint>

typedef __attribute__((ext_vector_type(16))) __bf16   v16bf;
typedef __attribute__((ext_vector_type(8)))  float    v8f;
typedef __attribute__((ext_vector_type(8)))  unsigned v8u;
typedef __attribute__((ext_vector_type(4)))  unsigned v4u;
typedef __attribute__((ext_vector_type(2)))  unsigned v2u;
typedef __attribute__((ext_vector_type(4)))  float    v4f;

#define BB 8
#define LL 1024
#define EE 512
#define HH 8
#define HDIM 64
#define PPOS 2047   // 2*L-1
#define LDSTR 36    // LDS tile row stride in u16 (72B: 8B-aligned, pad vs bank conflicts)

// ---------- helpers ----------
__device__ __forceinline__ unsigned short bf16r(float f) {
  unsigned u = __builtin_bit_cast(unsigned, f);
  u += 0x7FFFu + ((u >> 16) & 1u);            // round-to-nearest-even
  return (unsigned short)(u >> 16);
}
__device__ __forceinline__ unsigned packbf(float a, float b) {
  return (unsigned)bf16r(a) | ((unsigned)bf16r(b) << 16);
}
__device__ __forceinline__ v8f zero8f() {
  v8f z;
#pragma unroll
  for (int i = 0; i < 8; ++i) z[i] = 0.f;
  return z;
}
__device__ __forceinline__ v8u zero8u() {
  v8u z;
#pragma unroll
  for (int i = 0; i < 8; ++i) z[i] = 0u;
  return z;
}

// D = A(16x32 bf16) * B(32x16 bf16) + C(16x16 f32)
__device__ __forceinline__ v8f wmma_bf16(v8u a, v8u b, v8f c) {
  return __builtin_amdgcn_wmma_f32_16x16x32_bf16(
      false, __builtin_bit_cast(v16bf, a),
      false, __builtin_bit_cast(v16bf, b),
      (short)0, c, false, false);
}

// Build a 16x32 bf16 fragment from an LDS row (row-major [row][k], stride LDSTR u16).
// ISA layout: lane l -> row = l%16, khalf = (l/16)*8; VGPR r packs k = khalf+2r, +1;
// VGPR 4+r packs k = 16+khalf+2r, +1.
__device__ __forceinline__ v8u frag_lds(const unsigned short* row, int khalf) {
  const unsigned* p = (const unsigned*)row;   // dword view of bf16 pairs
  const int q0 = khalf >> 1;                  // 0 or 4
  v8u f;
#pragma unroll
  for (int r = 0; r < 4; ++r) { f[r] = p[q0 + r]; f[4 + r] = p[8 + q0 + r]; }
  return f;
}

// Fragment straight from global bf16 (rowptr = lane's matrix row, k-contiguous).
__device__ __forceinline__ v8u frag_g16(const unsigned short* rowptr, int kbase, int khalf) {
  v4u lo = *(const v4u*)(rowptr + kbase + khalf);
  v4u hi = *(const v4u*)(rowptr + kbase + 16 + khalf);
  v8u f;
#pragma unroll
  for (int r = 0; r < 4; ++r) { f[r] = lo[r]; f[4 + r] = hi[r]; }
  return f;
}

// Fragment from global f32 (convert to bf16 on the fly).
__device__ __forceinline__ v8u frag_g32(const float* rowptr, int kbase, int khalf) {
  v4f a0 = *(const v4f*)(rowptr + kbase + khalf);
  v4f a1 = *(const v4f*)(rowptr + kbase + khalf + 4);
  v4f b0 = *(const v4f*)(rowptr + kbase + 16 + khalf);
  v4f b1 = *(const v4f*)(rowptr + kbase + 16 + khalf + 4);
  v8u f;
  f[0] = packbf(a0[0], a0[1]); f[1] = packbf(a0[2], a0[3]);
  f[2] = packbf(a1[0], a1[1]); f[3] = packbf(a1[2], a1[3]);
  f[4] = packbf(b0[0], b0[1]); f[5] = packbf(b0[2], b0[3]);
  f[6] = packbf(b1[0], b1[1]); f[7] = packbf(b1[2], b1[3]);
  return f;
}

// ---------- kernel 1: generic C = A @ W^T (f32 in, bf16 WMMA, fused epilogues) ----------
// mode 0: QKV projection.  mode 1: positional projection.  mode 2: out projection (+bias).
__global__ __launch_bounds__(256) void rpmha_gemm(
    const float* __restrict__ A, const float* __restrict__ W,
    int M, int N, int K, int mode,
    const float* __restrict__ ub, const float* __restrict__ vbb,
    const float* __restrict__ bias,
    unsigned short* __restrict__ qu, unsigned short* __restrict__ qv,
    unsigned short* __restrict__ kk, unsigned short* __restrict__ vt,
    unsigned short* __restrict__ pk, float* __restrict__ outf) {
  __shared__ unsigned short As[128 * LDSTR];
  __shared__ unsigned short Bs[128 * LDSTR];
  const int n0 = blockIdx.x * 128;
  const int m0 = blockIdx.y * 128;
  const int t = threadIdx.x;
  const int wv = t >> 5, lane = t & 31;
  const int mw = (wv >> 2) * 64;   // wave row group
  const int nw = (wv & 3) * 32;    // wave col group
  const int khalf = (lane >> 4) * 8;
  const int rsel = lane & 15;

  v8f acc[4][2];
#pragma unroll
  for (int mi = 0; mi < 4; ++mi)
#pragma unroll
    for (int ni = 0; ni < 2; ++ni) acc[mi][ni] = zero8f();

  for (int k0 = 0; k0 < K; k0 += 32) {
    // Stage 128x32 tiles of A and W as bf16: float4 loads -> packed b64 LDS stores.
#pragma unroll
    for (int i = 0; i < 4; ++i) {
      int idx = (t + i * 256) * 4;        // element index in 128x32 tile
      int r = idx >> 5, c = idx & 31;     // c multiple of 4
      int m = m0 + r;
      v4f av;
      if (m < M) av = *(const v4f*)(A + (size_t)m * K + k0 + c);
      else { av[0] = 0.f; av[1] = 0.f; av[2] = 0.f; av[3] = 0.f; }
      v4f wv4 = *(const v4f*)(W + (size_t)(n0 + r) * K + k0 + c);
      v2u pa; pa[0] = packbf(av[0], av[1]);  pa[1] = packbf(av[2], av[3]);
      v2u pw; pw[0] = packbf(wv4[0], wv4[1]); pw[1] = packbf(wv4[2], wv4[3]);
      *(v2u*)&As[r * LDSTR + c] = pa;
      *(v2u*)&Bs[r * LDSTR + c] = pw;
    }
    __syncthreads();
    v8u af[4], bfr[2];
#pragma unroll
    for (int mi = 0; mi < 4; ++mi) af[mi] = frag_lds(&As[(mw + mi * 16 + rsel) * LDSTR], khalf);
#pragma unroll
    for (int ni = 0; ni < 2; ++ni) bfr[ni] = frag_lds(&Bs[(nw + ni * 16 + rsel) * LDSTR], khalf);
#pragma unroll
    for (int mi = 0; mi < 4; ++mi)
#pragma unroll
      for (int ni = 0; ni < 2; ++ni) acc[mi][ni] = wmma_bf16(af[mi], bfr[ni], acc[mi][ni]);
    __syncthreads();
  }

  const float scale = 0.044194173824159216f;  // 1/sqrt(E)
  const int mbase = m0 + mw + ((lane >> 4) << 3);  // C layout: M = r (+8 for upper half-wave)
#pragma unroll
  for (int mi = 0; mi < 4; ++mi) {
#pragma unroll
    for (int ni = 0; ni < 2; ++ni) {
#pragma unroll
      for (int r = 0; r < 8; ++r) {
        int m = mbase + mi * 16 + r;
        int n = n0 + nw + ni * 16 + rsel;
        float val = acc[mi][ni][r];
        if (m >= M) continue;
        if (mode == 0) {
          int b = m >> 10, i = m & 1023;
          int h = n / 192, c = n % 192, d = c & 63, which = c >> 6;
          size_t bh = (size_t)(b * HH + h);
          if (which == 0) {
            size_t idx = (bh * LL + i) * HDIM + d;
            qu[idx] = bf16r((val + ub[h * HDIM + d]) * scale);
            qv[idx] = bf16r((val + vbb[h * HDIM + d]) * scale);
          } else if (which == 1) {
            kk[(bh * LL + i) * HDIM + d] = bf16r(val);
          } else {
            vt[(bh * HDIM + d) * LL + i] = bf16r(val);   // V transposed: [b,h,d,j]
          }
        } else if (mode == 1) {
          int h = n >> 6, d = n & 63;
          pk[((size_t)h * PPOS + m) * HDIM + d] = bf16r(val);  // [h,p,d]
        } else {
          outf[(size_t)m * N + n] = val + bias[n];
        }
      }
    }
  }
}

// ---------- kernel 2: scores + rel_shift + softmax ----------
// One wave (32 threads) owns 16 query rows of one (b,h); full 16x1024 S lives in LDS,
// attn is written to HBM exactly once.
__global__ __launch_bounds__(32) void rpmha_scores(
    const unsigned short* __restrict__ qu, const unsigned short* __restrict__ qv,
    const unsigned short* __restrict__ kk, const unsigned short* __restrict__ pk,
    float* __restrict__ attn) {
  extern __shared__ float smem[];
  float* S = smem;                 // 16 x 1024
  float* bd = smem + 16 * 1024;    // 16 x 33 slab scratch (diagonal extraction)

  const int lane = threadIdx.x;
  const int bh = blockIdx.y;
  const int h = bh & 7;
  const int i0 = blockIdx.x * 16;
  const int khalf = (lane >> 4) * 8;
  const int rsel = lane & 15;
  const int iloc = ((lane >> 4) << 3);  // C-layout row offset (0 or 8)

  const unsigned short* quRow = qu + ((size_t)bh * LL + i0 + rsel) * HDIM;
  const unsigned short* qvRow = qv + ((size_t)bh * LL + i0 + rsel) * HDIM;
  v8u aqu[2], aqv[2];
#pragma unroll
  for (int s = 0; s < 2; ++s) {
    aqu[s] = frag_g16(quRow, s * 32, khalf);
    aqv[s] = frag_g16(qvRow, s * 32, khalf);
  }

  for (int j0 = 0; j0 < LL; j0 += 16) {
    // matrix_ac tile: (Qu) @ K^T  (K rows are d-contiguous -> direct B fragments)
    v8f acc = zero8f();
    const unsigned short* kRow = kk + ((size_t)bh * LL + j0 + rsel) * HDIM;
    if (j0 + 16 < LL) __builtin_prefetch(kRow + 16 * HDIM, 0, 1);
#pragma unroll
    for (int s = 0; s < 2; ++s) acc = wmma_bf16(aqu[s], frag_g16(kRow, s * 32, khalf), acc);

    // matrix_bd 16x32 slab at p_base = j0 - i0 + (L-1) - 15; shift picks the diagonal
    int pb = j0 - i0 + (LL - 1) - 15;
    v8f acc2[2];
#pragma unroll
    for (int tt = 0; tt < 2; ++tt) {
      acc2[tt] = zero8f();
      int p = pb + tt * 16 + rsel;
      v8u b0 = zero8u(), b1 = zero8u();
      if ((unsigned)p < (unsigned)PPOS) {
        const unsigned short* pRow = pk + ((size_t)h * PPOS + p) * HDIM;
        b0 = frag_g16(pRow, 0, khalf);
        b1 = frag_g16(pRow, 32, khalf);
      }
      acc2[tt] = wmma_bf16(aqv[0], b0, acc2[tt]);
      acc2[tt] = wmma_bf16(aqv[1], b1, acc2[tt]);
    }
    // dump bd slab to LDS, then combine along the rel-shift diagonal
#pragma unroll
    for (int tt = 0; tt < 2; ++tt)
#pragma unroll
      for (int r = 0; r < 8; ++r)
        bd[(iloc + r) * 33 + tt * 16 + rsel] = acc2[tt][r];
    __syncthreads();
#pragma unroll
    for (int r = 0; r < 8; ++r) {
      int il = iloc + r;
      int jl = rsel;
      S[il * 1024 + j0 + jl] = acc[r] + bd[il * 33 + (jl - il + 15)];
    }
    __syncthreads();
  }

  // softmax over each of the 16 rows (float4-vectorized), write normalized attn once
  for (int r = 0; r < 16; ++r) {
    float* Sr = S + r * 1024;
    float mx = -3.4e38f;
    for (int c = lane * 4; c < LL; c += 128) {
      v4f v = *(const v4f*)(Sr + c);
      mx = fmaxf(mx, fmaxf(fmaxf(v[0], v[1]), fmaxf(v[2], v[3])));
    }
#pragma unroll
    for (int off = 16; off > 0; off >>= 1) mx = fmaxf(mx, __shfl_xor(mx, off, 32));
    float sum = 0.f;
    for (int c = lane * 4; c < LL; c += 128) {
      v4f v = *(const v4f*)(Sr + c);
      v4f e;
      e[0] = __expf(v[0] - mx); e[1] = __expf(v[1] - mx);
      e[2] = __expf(v[2] - mx); e[3] = __expf(v[3] - mx);
      *(v4f*)(Sr + c) = e;
      sum += (e[0] + e[1]) + (e[2] + e[3]);
    }
#pragma unroll
    for (int off = 16; off > 0; off >>= 1) sum += __shfl_xor(sum, off, 32);
    float inv = 1.f / sum;
    float* dst = attn + ((size_t)bh * LL + i0 + r) * LL;
    for (int c = lane * 4; c < LL; c += 128) {
      v4f v = *(const v4f*)(Sr + c);
      v4f o; o[0] = v[0] * inv; o[1] = v[1] * inv; o[2] = v[2] * inv; o[3] = v[3] * inv;
      *(v4f*)(dst + c) = o;                 // global_store_b128, lane-contiguous
    }
  }
}

// ---------- kernel 3: ctx = attn @ V (attn f32 -> bf16 on the fly, Vt d-major) ----------
__global__ __launch_bounds__(256) void rpmha_ctx(
    const float* __restrict__ attn, const unsigned short* __restrict__ vt,
    float* __restrict__ ctx) {
  const int t = threadIdx.x, wv = t >> 5, lane = t & 31;
  const int bh = blockIdx.y, b = bh >> 3, h = bh & 7;
  const int i0 = blockIdx.x * 128 + wv * 16;
  const int khalf = (lane >> 4) * 8;
  const int rsel = lane & 15;

  const float* aRow = attn + ((size_t)bh * LL + i0 + rsel) * LL;
  v8f acc[4];
#pragma unroll
  for (int dt = 0; dt < 4; ++dt) acc[dt] = zero8f();

  for (int j0 = 0; j0 < LL; j0 += 32) {
    if (j0 + 256 < LL) __builtin_prefetch(aRow + j0 + 256, 0, 1);
    v8u af = frag_g32(aRow, j0, khalf);
#pragma unroll
    for (int dt = 0; dt < 4; ++dt) {
      const unsigned short* vRow = vt + ((size_t)bh * HDIM + dt * 16 + rsel) * LL;
      acc[dt] = wmma_bf16(af, frag_g16(vRow, j0, khalf), acc[dt]);
    }
  }
#pragma unroll
  for (int dt = 0; dt < 4; ++dt)
#pragma unroll
    for (int r = 0; r < 8; ++r) {
      int i = i0 + r + ((lane >> 4) << 3);
      int e = h * HDIM + dt * 16 + rsel;
      ctx[((size_t)b * LL + i) * EE + e] = acc[dt][r];
    }
}

// ---------- launch ----------
extern "C" void kernel_launch(void* const* d_in, const int* in_sizes, int n_in,
                              void* d_out, int out_size, void* d_ws, size_t ws_size,
                              hipStream_t stream) {
  const float* x     = (const float*)d_in[0];  // (B,L,E)
  const float* pose  = (const float*)d_in[1];  // (1,P,E)
  const float* in_w  = (const float*)d_in[2];  // (3E,E)
  const float* pos_w = (const float*)d_in[3];  // (E,E)
  const float* out_w = (const float*)d_in[4];  // (E,E)
  const float* out_b = (const float*)d_in[5];  // (E,)
  const float* ub    = (const float*)d_in[6];  // (HD,H) flat == (H,HD) flat
  const float* vb    = (const float*)d_in[7];

  float* out  = (float*)d_out;                       // (B,L,E)
  float* attn = out + (size_t)BB * LL * EE;          // (B,H,L,L)

  const size_t NQ = (size_t)BB * HH * LL * HDIM;     // 4,194,304
  unsigned short* qu = (unsigned short*)d_ws;
  unsigned short* qv = qu + NQ;
  unsigned short* kk = qv + NQ;
  unsigned short* vt = kk + NQ;
  unsigned short* pk = vt + NQ;                      // H*P*HD = 1,048,064
  float* ctxb = (float*)(pk + (size_t)HH * PPOS * HDIM);

  // 1) fused QKV projection + bias/scale/layout epilogue
  rpmha_gemm<<<dim3(12, 64), 256, 0, stream>>>(x, in_w, BB * LL, 3 * EE, EE, 0,
                                               ub, vb, nullptr,
                                               qu, qv, kk, vt, nullptr, nullptr);
  // 2) positional key projection
  rpmha_gemm<<<dim3(4, 16), 256, 0, stream>>>(pose, pos_w, PPOS, EE, EE, 1,
                                              nullptr, nullptr, nullptr,
                                              nullptr, nullptr, nullptr, nullptr, pk, nullptr);
  // 3) scores (ac + shifted bd) + softmax -> attn
  size_t smem = (size_t)(16 * 1024 + 16 * 33) * sizeof(float);
  rpmha_scores<<<dim3(64, 64), 32, smem, stream>>>(qu, qv, kk, pk, attn);
  // 4) context = attn @ V
  rpmha_ctx<<<dim3(8, 64), 256, 0, stream>>>(attn, vt, ctxb);
  // 5) output projection (+bias)
  rpmha_gemm<<<dim3(4, 64), 256, 0, stream>>>(ctxb, out_w, BB * LL, EE, EE, 2,
                                              nullptr, nullptr, out_b,
                                              nullptr, nullptr, nullptr, nullptr, nullptr, out);
}
